// GPT_20701742367221
// MI455X (gfx1250) — matrix operand order repaired and checked by reading, not compiled, not run on hardware
//
#include <hip/hip_runtime.h>
#include <math.h>

// ---------------------------------------------------------------------------
// GPT forward for MI455X (gfx1250, wave32). bf16 WMMA GEMMs with f32
// accumulate; LayerNorm / softmax / GELU / residual in f32.
// A-tiles staged via Tensor Data Mover (tensor_load_to_lds, TENSORcnt),
// double-buffered LDS, 8 WMMAs/wave per k-step.
// ---------------------------------------------------------------------------

typedef unsigned short bf16_t;
typedef unsigned int   u32;
typedef unsigned long long u64;

typedef __attribute__((ext_vector_type(16))) __bf16 v16bf;
typedef __attribute__((ext_vector_type(8)))  float  v8f;
typedef __attribute__((ext_vector_type(4)))  unsigned int v4u;
typedef __attribute__((ext_vector_type(8)))  int v8i;
typedef __attribute__((ext_vector_type(4)))  int v4i;

static __device__ __forceinline__ v8f v8f_zero() {
  v8f r;
#pragma unroll
  for (int i = 0; i < 8; ++i) r[i] = 0.0f;
  return r;
}

static __device__ __forceinline__ bf16_t f2bf(float x) {
  union { float f; u32 u; } c; c.f = x;
  u32 r = c.u + 0x7FFFu + ((c.u >> 16) & 1u);   // round-to-nearest-even
  return (bf16_t)(r >> 16);
}

union Frag {            // one 16x32 (A) or 32x16 (B) bf16 WMMA operand
  uint4 u4[2];
  v16bf v;
};

static __device__ __forceinline__ v8f wmma_bf16(v16bf a, v16bf b, v8f c) {
  return __builtin_amdgcn_wmma_f32_16x16x32_bf16(
      /*neg_a=*/false, a, /*neg_b=*/false, b,
      /*c_mod=*/(short)0, c, /*reuse_a=*/false, /*reuse_b=*/false);
}

// TDM: DMA a rows x 32 bf16 tile of a [Mtot,K] row-major tensor into LDS.
// D# layout per CDNA5 ISA ch.8 (group0: count/lds/global/type,
// group1: data_size, dims, tile dims, strides; remaining groups zero for 2D).
// This toolchain exposes the 6-arg builtin form.
static __device__ __forceinline__ void tdm_load_a_tile(
    const bf16_t* gsrc, u32 lds_off, int K, int Mtot, int rows) {
  const u64 ga = (u64)(const void*)gsrc;
  v4u g0;
  g0[0] = 1u;                                   // count=1, user descriptor
  g0[1] = lds_off;                              // LDS byte address
  g0[2] = (u32)(ga & 0xFFFFFFFFu);              // global_addr[31:0]
  g0[3] = (u32)(ga >> 32) | 0x80000000u;        // global_addr[56:32] | type=2
  v8i g1;
  g1[0] = 0x00010000;                           // data_size=1 -> 2 bytes
  g1[1] = (int)(((u32)K & 0xFFFFu) << 16);      // tensor_dim0[15:0]
  g1[2] = (int)((((u32)K >> 16) & 0xFFFFu) |
                (((u32)Mtot & 0xFFFFu) << 16)); // tensor_dim0[31:16]|dim1 lo
  g1[3] = (int)((((u32)Mtot >> 16) & 0xFFFFu) |
                (32u << 16));                   // dim1 hi | tile_dim0=32
  g1[4] = (int)((u32)rows & 0xFFFFu);           // tile_dim1=rows, tile_dim2=0
  g1[5] = (int)(u32)K;                          // tensor_dim0_stride[31:0]
  g1[6] = 0;                                    // stride0 hi | stride1 lo
  g1[7] = 0;
  v4i g2 = {0, 0, 0, 0};
  v4i g3 = {0, 0, 0, 0};
  v8i g4 = {0, 0, 0, 0, 0, 0, 0, 0};
  __builtin_amdgcn_tensor_load_to_lds(g0, g1, g2, g3, g4, 0);
}

// Model dims (compile-time)
constexpr int Bc = 4, Tc = 1024, Dc = 1024, Hc = 16, DHc = 64, FFc = 4096,
              Vc = 8192, Lc = 12;
constexpr int Mrows = Bc * Tc;   // 4096

// ---------------------------------------------------------------------------
// Embedding: x[b,t,:] = tk_emb[idx[b,t],:] + pos_emb[t,:]
// ---------------------------------------------------------------------------
__global__ __launch_bounds__(256) void embed_kernel(
    const int* __restrict__ idx, const float* __restrict__ tk,
    const float* __restrict__ pos, float* __restrict__ x) {
  const int row = blockIdx.x;            // b*T + t
  const int t = row & (Tc - 1);
  const int tok = idx[row];
  const int c = threadIdx.x * 4;
  float4 a = *(const float4*)(tk + (size_t)tok * Dc + c);
  float4 p = *(const float4*)(pos + (size_t)t * Dc + c);
  float4 r; r.x = a.x + p.x; r.y = a.y + p.y; r.z = a.z + p.z; r.w = a.w + p.w;
  *(float4*)(x + (size_t)row * Dc + c) = r;
}

// ---------------------------------------------------------------------------
// LayerNorm (f32 in) -> bf16 out.  One 256-thread block per row, D=1024.
// ---------------------------------------------------------------------------
__global__ __launch_bounds__(256) void ln_kernel(
    const float* __restrict__ x, const float* __restrict__ g,
    const float* __restrict__ be, bf16_t* __restrict__ out) {
  __shared__ float red[16];
  const int row = blockIdx.x;
  const int c = threadIdx.x * 4;
  const float4 v = *(const float4*)(x + (size_t)row * Dc + c);
  float s  = v.x + v.y + v.z + v.w;
  float ss = v.x * v.x + v.y * v.y + v.z * v.z + v.w * v.w;
#pragma unroll
  for (int off = 1; off < 32; off <<= 1) {
    s  += __shfl_xor(s, off);
    ss += __shfl_xor(ss, off);
  }
  const int wid = threadIdx.x >> 5, lane = threadIdx.x & 31;
  if (lane == 0) { red[wid] = s; red[8 + wid] = ss; }
  __syncthreads();
  float st = 0.f, sst = 0.f;
#pragma unroll
  for (int i = 0; i < 8; ++i) { st += red[i]; sst += red[8 + i]; }
  const float mean = st * (1.0f / Dc);
  const float var  = sst * (1.0f / Dc) - mean * mean;
  const float rstd = rsqrtf(var + 1e-5f);
  const float4 gv = *(const float4*)(g + c);
  const float4 bv = *(const float4*)(be + c);
  const u32 y0 = f2bf((v.x - mean) * rstd * gv.x + bv.x);
  const u32 y1 = f2bf((v.y - mean) * rstd * gv.y + bv.y);
  const u32 y2 = f2bf((v.z - mean) * rstd * gv.z + bv.z);
  const u32 y3 = f2bf((v.w - mean) * rstd * gv.w + bv.w);
  uint2 pk; pk.x = y0 | (y1 << 16); pk.y = y2 | (y3 << 16);
  *(uint2*)(out + (size_t)row * Dc + c) = pk;
}

// ---------------------------------------------------------------------------
// WMMA GEMM: C = A(bf16,[M,K] row-major) * B(f32,[K,N]) + bias
//   BM=128, BN=64, BK=32. 128 threads (4 waves); wave computes 32x64
//   (2x4 WMMA tiles = 8 v_wmma per k-step). A tile via TDM, double-buffered.
// HEADED: B is per-head [H][K][64] (Wq/Wk/Wv); BN=64 tile == one head.
// Output modes:
//   0 OUT_BF16_ROW : bf16 [M,N] row-major  (optionally GELU)
//   1 OUT_BHTD     : bf16 scatter to [B,H,T,DH]   (Q, K)
//   2 OUT_BHDT     : bf16 scatter to [B,H,DH,T]   (V transposed)
//   3 OUT_F32_RESID: f32  out[m,n] += acc         (residual stream)
//   4 OUT_F32_ROW  : f32  [M,N] row-major         (logits)
// ---------------------------------------------------------------------------
constexpr int BM = 128, BN = 64, BK = 32;

template <bool HEADED, int OM, bool DO_GELU>
__global__ __launch_bounds__(128) void gemm_kernel(
    const bf16_t* __restrict__ A, const float* __restrict__ Bw,
    const float* __restrict__ bias, void* __restrict__ outp,
    int M, int N, int K) {
  __shared__ bf16_t lsA[2][BM * BK];    // [m][k]  (TDM destination)
  __shared__ bf16_t lsB[2][BN * BK];    // [n][k]  (k-transposed)
  const int n0 = blockIdx.x * BN;
  const int m0 = blockIdx.y * BM;
  const int tid = threadIdx.x;
  const int lane = tid & 31;
  const int w = tid >> 5;
  const int wm = w * 32;               // wave row offset in tile
  const int nl = lane & 15;
  const int half = lane >> 4;

  const float* bptr = HEADED ? (Bw + (size_t)(n0 >> 6) * (size_t)K * 64) : Bw;
  const int ldb = HEADED ? 64 : N;
  const int ncb = HEADED ? 0 : n0;
  const bf16_t* atile = A + (size_t)m0 * K;

  v8f acc[2][4];
#pragma unroll
  for (int im = 0; im < 2; ++im)
#pragma unroll
    for (int in_ = 0; in_ < 4; ++in_) acc[im][in_] = v8f_zero();

  // B cooperative-load assignment (32x64 f32 -> bf16 [n][k])
  const int bn4 = (tid & 15) * 4;      // col group
  const int bk = tid >> 4;             // base k row 0..7

  auto load_b = [&](int k0, int bufi) {
#pragma unroll
    for (int sIt = 0; sIt < 4; ++sIt) {
      const int kk = bk + sIt * 8;
      const float4 f =
          *(const float4*)(bptr + (size_t)(k0 + kk) * ldb + ncb + bn4);
      bf16_t* dst = &lsB[bufi][0];
      dst[(bn4 + 0) * BK + kk] = f2bf(f.x);
      dst[(bn4 + 1) * BK + kk] = f2bf(f.y);
      dst[(bn4 + 2) * BK + kk] = f2bf(f.z);
      dst[(bn4 + 3) * BK + kk] = f2bf(f.w);
    }
  };

  // ---- prologue: stage tile k0=0 into buffer 0
  if (w == 0)
    tdm_load_a_tile(atile, (u32)(u64)(const void*)&lsA[0][0], K, M, BM);
  load_b(0, 0);
  if (w == 0) __builtin_amdgcn_s_wait_tensorcnt(0);
  __syncthreads();

  int buf = 0;
  for (int k0 = 0; k0 < K; k0 += BK) {
    const int kn = k0 + BK;
    if (kn < K) {                      // stage next tile while computing
      if (w == 0)
        tdm_load_a_tile(atile + kn, (u32)(u64)(const void*)&lsA[buf ^ 1][0],
                        K, M, BM);
      load_b(kn, buf ^ 1);
    }

    Frag a[2], b[4];
    const uint4* pa = (const uint4*)&lsA[buf][0];  // 4 uint4 per 32-elem row
    const uint4* pb = (const uint4*)&lsB[buf][0];
#pragma unroll
    for (int im = 0; im < 2; ++im) {
      const int row = wm + im * 16 + nl;    // A frag: lane row, K striped
      a[im].u4[0] = pa[row * 4 + 2 * half];
      a[im].u4[1] = pa[row * 4 + 2 * half + 1];
    }
#pragma unroll
    for (int in_ = 0; in_ < 4; ++in_) {
      const int col = in_ * 16 + nl;        // B frag: lane col, K striped
      b[in_].u4[0] = pb[col * 4 + 2 * half];
      b[in_].u4[1] = pb[col * 4 + 2 * half + 1];
    }
#pragma unroll
    for (int im = 0; im < 2; ++im)
#pragma unroll
      for (int in_ = 0; in_ < 4; ++in_)
        acc[im][in_] = wmma_bf16(a[im].v, b[in_].v, acc[im][in_]);

    if (kn < K && w == 0) __builtin_amdgcn_s_wait_tensorcnt(0);
    __syncthreads();
    buf ^= 1;
  }

  // Epilogue. C layout: lane nl = col, row = i + 8*half for VGPR i.
#pragma unroll
  for (int im = 0; im < 2; ++im) {
#pragma unroll
    for (int in_ = 0; in_ < 4; ++in_) {
      const int col = n0 + in_ * 16 + nl;
      const float bv = bias[col];
#pragma unroll
      for (int i = 0; i < 8; ++i) {
        const int row = m0 + wm + im * 16 + i + 8 * half;
        float val = acc[im][in_][i] + bv;
        if constexpr (DO_GELU)
          val = 0.5f * val * (1.0f + erff(val * 0.70710678118654752f));
        if constexpr (OM == 0) {
          ((bf16_t*)outp)[(size_t)row * N + col] = f2bf(val);
        } else if constexpr (OM == 3) {
          float* o = (float*)outp;
          const size_t ix = (size_t)row * N + col;
          o[ix] = o[ix] + val;
        } else if constexpr (OM == 4) {
          ((float*)outp)[(size_t)row * N + col] = val;
        } else if constexpr (OM == 1) {   // [B,H,T,DH]
          const int bi = row >> 10, t = row & (Tc - 1);
          const int h = col >> 6, e = col & 63;
          ((bf16_t*)outp)[(((size_t)(bi * Hc + h)) * Tc + t) * DHc + e] =
              f2bf(val);
        } else {                          // [B,H,DH,T]
          const int bi = row >> 10, t = row & (Tc - 1);
          const int h = col >> 6, e = col & 63;
          ((bf16_t*)outp)[(((size_t)(bi * Hc + h)) * DHc + e) * Tc + t] =
              f2bf(val);
        }
      }
    }
  }
}

// ---------------------------------------------------------------------------
// Flash attention.  grid(T/64, H, B), 128 threads (4 waves).
// Wave owns a 16-row Q tile; streams causal 32-col S chunks.
// q,k: bf16 [B,H,T,64]; vT: bf16 [B,H,64,T]; out: bf16 [B,T,D].
// ---------------------------------------------------------------------------
__global__ __launch_bounds__(128) void attn_kernel(
    const bf16_t* __restrict__ q, const bf16_t* __restrict__ k,
    const bf16_t* __restrict__ vT, bf16_t* __restrict__ o) {
  __shared__ bf16_t lsP[4][16 * 32];     // per-wave P staging
  const int w = threadIdx.x >> 5, lane = threadIdx.x & 31;
  const int nl = lane & 15, half = lane >> 4;
  const int b = blockIdx.z, h = blockIdx.y;
  const int q0 = blockIdx.x * 64 + w * 16;
  const size_t bh = (size_t)(b * Hc + h);
  const uint4* qp = (const uint4*)(q + bh * Tc * DHc);
  const uint4* kp = (const uint4*)(k + bh * Tc * DHc);
  const uint4* vp = (const uint4*)(vT + bh * DHc * Tc);

  // Q fragments (A layout) for dh chunks 0..31 / 32..63, held in registers.
  Frag Qf[2];
  const int tq = q0 + nl;
#pragma unroll
  for (int c = 0; c < 2; ++c) {
    Qf[c].u4[0] = qp[tq * 8 + 4 * c + 2 * half];
    Qf[c].u4[1] = qp[tq * 8 + 4 * c + 2 * half + 1];
  }

  v8f oacc[4];
#pragma unroll
  for (int nt = 0; nt < 4; ++nt) oacc[nt] = v8f_zero();
  float mrow[8], lrow[8];
#pragma unroll
  for (int i = 0; i < 8; ++i) { mrow[i] = -1e30f; lrow[i] = 0.0f; }

  for (int s0 = 0; s0 <= q0 + 15; s0 += 32) {
    // ---- scores: two 16x16 tiles, each = 2 WMMAs over DH=64
    v8f sacc[2];
#pragma unroll
    for (int st = 0; st < 2; ++st) {
      const int s = s0 + st * 16 + nl;   // B frag: lane col = key index
      Frag kf0, kf1;
      kf0.u4[0] = kp[s * 8 + 2 * half];
      kf0.u4[1] = kp[s * 8 + 2 * half + 1];
      kf1.u4[0] = kp[s * 8 + 4 + 2 * half];
      kf1.u4[1] = kp[s * 8 + 4 + 2 * half + 1];
      v8f z = v8f_zero();
      z = wmma_bf16(Qf[0].v, kf0.v, z);
      z = wmma_bf16(Qf[1].v, kf1.v, z);
      sacc[st] = z;
    }
    const bool need_mask = (s0 + 31) > q0;

    // ---- online softmax (per-row stats; row = i + 8*half)
#pragma unroll
    for (int i = 0; i < 8; ++i) {
      float a0 = sacc[0][i] * 0.125f;    // 1/sqrt(64)
      float a1 = sacc[1][i] * 0.125f;
      if (need_mask) {
        const int trow = q0 + i + 8 * half;
        if (s0 + nl > trow)      a0 = -1e30f;
        if (s0 + 16 + nl > trow) a1 = -1e30f;
      }
      float mx = fmaxf(a0, a1);
      mx = fmaxf(mx, __shfl_xor(mx, 1));
      mx = fmaxf(mx, __shfl_xor(mx, 2));
      mx = fmaxf(mx, __shfl_xor(mx, 4));
      mx = fmaxf(mx, __shfl_xor(mx, 8));
      const float mnew = fmaxf(mrow[i], mx);
      const float fsc = __expf(mrow[i] - mnew);
      const float p0 = __expf(a0 - mnew);
      const float p1 = __expf(a1 - mnew);
      float rs = p0 + p1;
      rs += __shfl_xor(rs, 1);
      rs += __shfl_xor(rs, 2);
      rs += __shfl_xor(rs, 4);
      rs += __shfl_xor(rs, 8);
      lrow[i] = lrow[i] * fsc + rs;
      mrow[i] = mnew;
#pragma unroll
      for (int nt = 0; nt < 4; ++nt) oacc[nt][i] *= fsc;
      const int m = i + 8 * half;
      lsP[w][m * 32 + nl]      = f2bf(p0);
      lsP[w][m * 32 + 16 + nl] = f2bf(p1);
    }
    asm volatile("s_wait_dscnt 0" ::: "memory");   // wave-local LDS RAW

    // ---- P (A layout) x V chunk (B layout, contiguous in t via vT)
    Frag pf;
    const uint4* pp = (const uint4*)lsP[w];
    pf.u4[0] = pp[nl * 4 + 2 * half];
    pf.u4[1] = pp[nl * 4 + 2 * half + 1];
#pragma unroll
    for (int nt = 0; nt < 4; ++nt) {
      const int dh = nt * 16 + nl;
      const int base = dh * (Tc / 8) + (s0 >> 3) + 2 * half;
      Frag vf;
      vf.u4[0] = vp[base];
      vf.u4[1] = vp[base + 1];
      oacc[nt] = wmma_bf16(pf.v, vf.v, oacc[nt]);
    }
  }

  // ---- normalize + write [B,T,D] bf16
#pragma unroll
  for (int i = 0; i < 8; ++i) {
    const float inv = 1.0f / lrow[i];
    const int t = q0 + i + 8 * half;
    const size_t rb = ((size_t)(b * Tc + t)) * Dc + h * DHc;
#pragma unroll
    for (int nt = 0; nt < 4; ++nt)
      o[rb + nt * 16 + nl] = f2bf(oacc[nt][i] * inv);
  }
}

// ---------------------------------------------------------------------------
// Host launcher
// ---------------------------------------------------------------------------
extern "C" void kernel_launch(void* const* d_in, const int* in_sizes, int n_in,
                              void* d_out, int out_size, void* d_ws,
                              size_t ws_size, hipStream_t stream) {
  (void)in_sizes; (void)n_in; (void)out_size; (void)ws_size;
  const int*   idx     = (const int*)d_in[0];
  const float* tk_emb  = (const float*)d_in[1];
  const float* pos_emb = (const float*)d_in[2];
  const float* ln1_g   = (const float*)d_in[3];
  const float* ln1_b   = (const float*)d_in[4];
  const float* Wq      = (const float*)d_in[5];
  const float* bq      = (const float*)d_in[6];
  const float* Wk      = (const float*)d_in[7];
  const float* bk      = (const float*)d_in[8];
  const float* Wv      = (const float*)d_in[9];
  const float* bv      = (const float*)d_in[10];
  const float* Wo      = (const float*)d_in[11];
  const float* bo      = (const float*)d_in[12];
  const float* ln2_g   = (const float*)d_in[13];
  const float* ln2_b   = (const float*)d_in[14];
  const float* W1      = (const float*)d_in[15];
  const float* b1      = (const float*)d_in[16];
  const float* W2      = (const float*)d_in[17];
  const float* b2      = (const float*)d_in[18];
  const float* lnf_g   = (const float*)d_in[19];
  const float* lnf_b   = (const float*)d_in[20];
  const float* Wout    = (const float*)d_in[21];
  const float* bout    = (const float*)d_in[22];
  float* out = (float*)d_out;

  // Workspace carve-up (~88 MB)
  char* wsp = (char*)d_ws;
  auto carve = [&](size_t bytes) {
    char* p = wsp;
    wsp += (bytes + 255) & ~size_t(255);
    return p;
  };
  float*  x   = (float*)carve((size_t)Mrows * Dc * 4);   // residual stream
  bf16_t* hbf = (bf16_t*)carve((size_t)Mrows * Dc * 2);  // LN output
  bf16_t* qb  = (bf16_t*)carve((size_t)Mrows * Dc * 2);  // Q [B,H,T,DH]
  bf16_t* kb  = (bf16_t*)carve((size_t)Mrows * Dc * 2);  // K [B,H,T,DH]
  bf16_t* vb  = (bf16_t*)carve((size_t)Mrows * Dc * 2);  // V [B,H,DH,T]
  bf16_t* ob  = (bf16_t*)carve((size_t)Mrows * Dc * 2);  // attn out [B,T,D]
  bf16_t* ffb = (bf16_t*)carve((size_t)Mrows * FFc * 2); // GELU(FF1)

  embed_kernel<<<Mrows, 256, 0, stream>>>(idx, tk_emb, pos_emb, x);

  const dim3 gD(Dc / BN, Mrows / BM);    // N=1024 GEMMs
  const dim3 gF(FFc / BN, Mrows / BM);   // N=4096 GEMM
  const dim3 gV(Vc / BN, Mrows / BM);    // N=8192 GEMM

  for (int l = 0; l < Lc; ++l) {
    const float* wq = Wq + (size_t)l * Hc * Dc * DHc;
    const float* wk = Wk + (size_t)l * Hc * Dc * DHc;
    const float* wv = Wv + (size_t)l * Hc * Dc * DHc;
    const float* wo = Wo + (size_t)l * Dc * Dc;
    const float* w1 = W1 + (size_t)l * Dc * FFc;
    const float* w2 = W2 + (size_t)l * FFc * Dc;
    const float* bql = bq + (size_t)l * Hc * DHc;
    const float* bkl = bk + (size_t)l * Hc * DHc;
    const float* bvl = bv + (size_t)l * Hc * DHc;
    const float* bol = bo + (size_t)l * Dc;
    const float* b1l = b1 + (size_t)l * FFc;
    const float* b2l = b2 + (size_t)l * Dc;

    ln_kernel<<<Mrows, 256, 0, stream>>>(x, ln1_g + l * Dc, ln1_b + l * Dc,
                                         hbf);
    gemm_kernel<true, 1, false><<<gD, 128, 0, stream>>>(hbf, wq, bql, qb,
                                                        Mrows, Dc, Dc);
    gemm_kernel<true, 1, false><<<gD, 128, 0, stream>>>(hbf, wk, bkl, kb,
                                                        Mrows, Dc, Dc);
    gemm_kernel<true, 2, false><<<gD, 128, 0, stream>>>(hbf, wv, bvl, vb,
                                                        Mrows, Dc, Dc);
    attn_kernel<<<dim3(Tc / 64, Hc, Bc), 128, 0, stream>>>(qb, kb, vb, ob);
    gemm_kernel<false, 3, false><<<gD, 128, 0, stream>>>(ob, wo, bol, x,
                                                         Mrows, Dc, Dc);
    ln_kernel<<<Mrows, 256, 0, stream>>>(x, ln2_g + l * Dc, ln2_b + l * Dc,
                                         hbf);
    gemm_kernel<false, 0, true><<<gF, 128, 0, stream>>>(hbf, w1, b1l, ffb,
                                                        Mrows, FFc, Dc);
    gemm_kernel<false, 3, false><<<gD, 128, 0, stream>>>(ffb, w2, b2l, x,
                                                         Mrows, Dc, FFc);
  }

  ln_kernel<<<Mrows, 256, 0, stream>>>(x, lnf_g, lnf_b, hbf);
  gemm_kernel<false, 4, false><<<gV, 128, 0, stream>>>(hbf, Wout, bout, out,
                                                       Mrows, Vc, Dc);
}
